// InformerEncoderLayer_35347580846624
// MI455X (gfx1250) — matrix operand order repaired and checked by reading, not compiled, not run on hardware
//
#include <hip/hip_runtime.h>

// Informer encoder layer for MI455X (gfx1250, wave32, WMMA).
// B=4, S=4096, D=128, DFF=256, top-k = int(0.02*4096) = 81.
//
// prep_kernel : x -> bf16 hi/lo split buffers; W1/W2 -> transposed f32 copies.
// attn_ln1    : scores via bf16-split WMMA (hi*hi+hi*lo+lo*hi, f32 accum),
//               exact top-81 radix select in 320KB WGP LDS, softmax, sparse
//               A*V, residual + LayerNorm1 -> h.
// ffn_ln2     : both FFN GEMMs as V_WMMA_F32_16X16X4_F32 tiles (exact f32),
//               fused bias/relu/residual/LayerNorm2.

#define SEQ   4096
#define DM    128
#define DFF   256
#define BATCH 4
#define TOPK  81
#define NWAVE 16
#define BLOCK1 512
#define BLOCK2 512
#define MAXK   128

typedef __attribute__((ext_vector_type(16))) __bf16 v16bf;
typedef __attribute__((ext_vector_type(8)))  __bf16 v8bf;
typedef __attribute__((ext_vector_type(8)))  float  v8f;
typedef __attribute__((ext_vector_type(2)))  float  v2f;

// Monotonic unsigned key for f32 ordering (IEEE trick).
__device__ __forceinline__ unsigned ordkey(float s) {
    unsigned u = __float_as_uint(s);
    return (u & 0x80000000u) ? ~u : (u | 0x80000000u);
}
__device__ __forceinline__ float wave_sum(float v) {
#pragma unroll
    for (int m = 16; m >= 1; m >>= 1) v += __shfl_xor(v, m, 32);
    return v;
}
__device__ __forceinline__ float wave_max(float v) {
#pragma unroll
    for (int m = 16; m >= 1; m >>= 1) v = fmaxf(v, __shfl_xor(v, m, 32));
    return v;
}

// ---------------------------------------------------------------------------
// Kernel 0: one-time data prep.
//  - split x (f32) into bf16 hi/lo so kernel 1's WMMA operands are straight
//    vector loads (no per-tile cvt chains in the hot loop).
//  - transpose W1 -> w1t[n][k] (256x128), W2 -> w2t[n][e] (128x256) so the
//    f32-WMMA B fragments in kernel 2 are contiguous 8B loads.
// ---------------------------------------------------------------------------
__global__ __launch_bounds__(256, 1)
void prep_kernel(const float* __restrict__ x,
                 const float* __restrict__ W1,
                 const float* __restrict__ W2,
                 __bf16* __restrict__ xhi,
                 __bf16* __restrict__ xlo,
                 float* __restrict__ w1t,
                 float* __restrict__ w2t)
{
    const int i = blockIdx.x * 256 + threadIdx.x;
    const int total = BATCH * SEQ * DM;            // 2,097,152
    if (i < total) {
        float f = x[i];
        __bf16 h = (__bf16)f;
        xhi[i] = h;
        xlo[i] = (__bf16)(f - (float)h);
    }
    if (i < DM * DFF) {                            // 32,768
        int k = i / DFF, n = i % DFF;              // W1[k][n]
        w1t[n * DM + k] = W1[i];
        int e = i / DM, n2 = i % DM;               // W2[e][n2]
        w2t[n2 * DFF + e] = W2[i];
    }
}

// ---------------------------------------------------------------------------
// Kernel 1: prob-sparse attention + residual + LN1.
// One block = 16 query rows of one batch; 16 waves. Dynamic LDS ~295 KB.
// ---------------------------------------------------------------------------
__global__ __launch_bounds__(BLOCK1, 1)
void attn_ln1_kernel(const float* __restrict__ x,
                     const __bf16* __restrict__ xhi,
                     const __bf16* __restrict__ xlo,
                     const float* __restrict__ g1,
                     const float* __restrict__ bln1,
                     float* __restrict__ hout)
{
    extern __shared__ __align__(32) unsigned char smem[];
    float*    sc    = (float*)(smem);                                   // 16*4096 f32
    unsigned* hist  = (unsigned*)(smem + (size_t)16*SEQ*4);             // 16*256
    int*      lidx  = (int*)(smem + (size_t)16*SEQ*4 + 16*256*4);       // 16*128
    float*    lw    = (float*)(smem + (size_t)16*SEQ*4 + 16*256*4 + 16*MAXK*4); // 16*128
    unsigned* rpref = (unsigned*)(smem + (size_t)16*SEQ*4 + 16*256*4 + 16*MAXK*8);
    int*      rk    = (int*)(smem + (size_t)16*SEQ*4 + 16*256*4 + 16*MAXK*8 + 16*4);

    const int tid  = threadIdx.x;
    const int lane = tid & 31;
    const int wave = tid >> 5;
    const int half = lane >> 4;
    const int mn   = lane & 15;
    const int bidx = blockIdx.y;
    const int q0   = blockIdx.x * 16;
    const float*  X  = x   + (size_t)bidx * SEQ * DM;
    const __bf16* XH = xhi + (size_t)bidx * SEQ * DM;
    const __bf16* XL = xlo + (size_t)bidx * SEQ * DM;

    // ---- A fragments (Q tile): 16-bit A 16x32 layout -> two contiguous
    //      8-element runs per chunk: cols [32c+8h, +8) and [32c+16+8h, +8).
    v16bf qhi[4], qlo[4];
    {
        const __bf16* qh = XH + (size_t)(q0 + mn) * DM;
        const __bf16* ql = XL + (size_t)(q0 + mn) * DM;
#pragma unroll
        for (int c = 0; c < 4; ++c) {
            v8bf h0 = *(const v8bf*)(qh + 32*c + half*8);
            v8bf h1 = *(const v8bf*)(qh + 32*c + 16 + half*8);
            v8bf l0 = *(const v8bf*)(ql + 32*c + half*8);
            v8bf l1 = *(const v8bf*)(ql + 32*c + 16 + half*8);
            qhi[c] = __builtin_shufflevector(h0, h1, 0,1,2,3,4,5,6,7,8,9,10,11,12,13,14,15);
            qlo[c] = __builtin_shufflevector(l0, l1, 0,1,2,3,4,5,6,7,8,9,10,11,12,13,14,15);
        }
    }

    const float SCALE = 0.08838834764831845f;   // 1/sqrt(128)

    // ---- scores: each wave computes 16 disjoint 16x16 key tiles.
    // 16-bit B 32x16 layout -> two contiguous 8-element runs per chunk:
    // cols [32c+8h, +8) and [32c+16+8h, +8) of key row (key0 + lane%16).
    for (int kt = wave; kt < SEQ/16; kt += NWAVE) {
        const int key0 = kt * 16;
        const __bf16* kh = XH + (size_t)(key0 + mn) * DM;
        const __bf16* kl = XL + (size_t)(key0 + mn) * DM;
        v8f acc = {};
#pragma unroll
        for (int c = 0; c < 4; ++c) {
            v16bf bhi = __builtin_shufflevector(*(const v8bf*)(kh + 32*c + half*8), *(const v8bf*)(kh + 32*c + 16 + half*8), 0,1,2,3,4,5,6,7,8,9,10,11,12,13,14,15);
            v16bf blo = __builtin_shufflevector(*(const v8bf*)(kl + 32*c + half*8), *(const v8bf*)(kl + 32*c + 16 + half*8), 0,1,2,3,4,5,6,7,8,9,10,11,12,13,14,15);
            acc = __builtin_amdgcn_wmma_f32_16x16x32_bf16(false, qhi[c], false, bhi,
                                                          (short)0, acc, false, false);
            acc = __builtin_amdgcn_wmma_f32_16x16x32_bf16(false, qhi[c], false, blo,
                                                          (short)0, acc, false, false);
            acc = __builtin_amdgcn_wmma_f32_16x16x32_bf16(false, qlo[c], false, bhi,
                                                          (short)0, acc, false, false);
        }
#pragma unroll
        for (int r = 0; r < 8; ++r) {
            int m = r + 8*half;                    // C: VGPR r -> M=r+8*(l/16)
            sc[(size_t)m*SEQ + key0 + mn] = acc[r] * SCALE;
        }
    }
    __syncthreads();

    // ---- wave w owns row q0+w: exact 81st-largest via 4-pass radix select.
    float* row = sc + (size_t)wave * SEQ;

    float mx = -3.4e38f;
    for (int i = lane; i < SEQ; i += 32) mx = fmaxf(mx, row[i]);
    mx = wave_max(mx);

    for (int pass = 0; pass < 4; ++pass) {
        for (int j = lane; j < 256; j += 32) hist[wave*256 + j] = 0u;
        __syncthreads();
        const int shift = 24 - 8*pass;
        const unsigned pref = (pass == 0) ? 0u : rpref[wave];
        for (int i = lane; i < SEQ; i += 32) {
            unsigned u = ordkey(row[i]);
            bool inc = (pass == 0) || ((u >> (shift + 8)) == (pref >> (shift + 8)));
            if (inc) atomicAdd(&hist[wave*256 + ((u >> shift) & 255u)], 1u);
        }
        __syncthreads();
        if (lane == 0) {
            int kneed = (pass == 0) ? TOPK : rk[wave];
            unsigned cum = 0, sel = 0;
            for (int d = 255; d >= 0; --d) {
                unsigned c = hist[wave*256 + d];
                if (cum + c >= (unsigned)kneed) { sel = (unsigned)d; kneed -= (int)cum; break; }
                cum += c;
            }
            rpref[wave] = ((pass == 0) ? 0u : rpref[wave]) | (sel << shift);
            rk[wave] = kneed;
        }
        __syncthreads();
    }
    const unsigned T = rpref[wave];

    // ---- deterministic ballot-scan compaction of the selected set.
    float dsum = 0.0f;
    int base = 0;
    for (int i0 = 0; i0 < SEQ; i0 += 32) {
        int i = i0 + lane;
        float s = row[i];
        bool sel = (ordkey(s) >= T);
        unsigned m32 = (unsigned)__ballot(sel);
        int pos = base + __popc(m32 & ((1u << lane) - 1u));
        if (sel && pos < MAXK) {
            float w = __expf(s - mx);
            lidx[wave*MAXK + pos] = i;
            lw[wave*MAXK + pos]   = w;
            dsum += w;
        }
        base += __popc(m32);
    }
    const float denom = wave_sum(dsum);
    const int n = (base < MAXK) ? base : MAXK;
    __syncthreads();

    // ---- sparse A*V (lane owns 4 dims) + residual + LN1.
    const int col = lane * 4;
    float a0 = 0.f, a1 = 0.f, a2 = 0.f, a3 = 0.f;
    for (int p = 0; p < n; ++p) {
        int idx = lidx[wave*MAXK + p];
        float w = lw[wave*MAXK + p];
        const float* vp = X + (size_t)idx * DM + col;
        a0 += w * vp[0]; a1 += w * vp[1]; a2 += w * vp[2]; a3 += w * vp[3];
    }
    const float inv = 1.0f / denom;
    const float* xp = X + (size_t)(q0 + wave) * DM + col;
    float y0 = xp[0] + a0*inv, y1 = xp[1] + a1*inv;
    float y2 = xp[2] + a2*inv, y3 = xp[3] + a3*inv;

    float mu = wave_sum(y0 + y1 + y2 + y3) * (1.0f/128.0f);
    float d0 = y0 - mu, d1 = y1 - mu, d2 = y2 - mu, d3 = y3 - mu;
    float var = wave_sum(d0*d0 + d1*d1 + d2*d2 + d3*d3) * (1.0f/128.0f);
    float rs = __frsqrt_rn(var + 1e-5f);

    float* hp = hout + (size_t)(bidx * SEQ + q0 + wave) * DM + col;
    hp[0] = d0*rs*g1[col+0] + bln1[col+0];
    hp[1] = d1*rs*g1[col+1] + bln1[col+1];
    hp[2] = d2*rs*g1[col+2] + bln1[col+2];
    hp[3] = d3*rs*g1[col+3] + bln1[col+3];
}

// ---------------------------------------------------------------------------
// Kernel 2: ff = relu(h@W1+b1)@W2+b2; out = LN2(h + ff).
// 128 rows per block; both GEMMs on V_WMMA_F32_16X16X4_F32 (exact f32).
// LDS: h block 64KB + ff1 128KB + y 64KB = 256KB.
// f32 A 16x4 layout: lane l -> M=l%16, K = 2*(l/16)+v  => chunk c is a v2f
// at element offset 4c + 2*(l/16). B mirrors with N=l%16 (uses transposed W).
// ---------------------------------------------------------------------------
__global__ __launch_bounds__(BLOCK2, 1)
void ffn_ln2_kernel(const float* __restrict__ h,
                    const float* __restrict__ w1t,
                    const float* __restrict__ B1,
                    const float* __restrict__ w2t,
                    const float* __restrict__ B2,
                    const float* __restrict__ g2,
                    const float* __restrict__ bln2,
                    float* __restrict__ out)
{
    extern __shared__ __align__(16) unsigned char smem[];
    float* hl  = (float*)smem;               // 128*128
    float* ff1 = hl + 128*DM;                // 128*256
    float* yl  = ff1 + 128*DFF;              // 128*128

    const int tid = threadIdx.x, lane = tid & 31, wave = tid >> 5;
    const int half = lane >> 4, mn = lane & 15;
    const size_t row0 = (size_t)blockIdx.x * 128;

    for (int i = tid; i < 128*DM; i += BLOCK2) hl[i] = h[row0*DM + i];
    __syncthreads();

    const int mtile = wave >> 1;             // 8 M-tiles of 16 rows
    const int ng    = wave & 1;

    // ---- layer 1: ff1 = relu(h @ W1 + b1). 8x16 tiles, 2 waves per M-tile.
    {
        v2f areg[32];
#pragma unroll
        for (int c = 0; c < 32; ++c)
            areg[c] = *(const v2f*)(hl + (mtile*16 + mn)*DM + 4*c + 2*half);
        for (int t = 0; t < 8; ++t) {
            const int n0 = (ng*8 + t) * 16;
            v8f acc = {};
#pragma unroll
            for (int c = 0; c < 32; ++c) {
                v2f bfrag = *(const v2f*)(w1t + (size_t)(n0 + mn)*DM + 4*c + 2*half);
                acc = __builtin_amdgcn_wmma_f32_16x16x4_f32(false, areg[c], false, bfrag,
                                                            (short)0, acc, false, false);
            }
            const float bias = B1[n0 + mn];
#pragma unroll
            for (int r = 0; r < 8; ++r) {
                int m = mtile*16 + r + 8*half;
                ff1[m*DFF + n0 + mn] = fmaxf(acc[r] + bias, 0.0f);
            }
        }
    }
    __syncthreads();

    // ---- layer 2: y = h + ff1 @ W2 + b2. 8x8 tiles, 2 waves per M-tile.
    {
        v2f areg[64];
#pragma unroll
        for (int c = 0; c < 64; ++c)
            areg[c] = *(const v2f*)(ff1 + (mtile*16 + mn)*DFF + 4*c + 2*half);
        for (int t = 0; t < 4; ++t) {
            const int n0 = (ng*4 + t) * 16;
            v8f acc = {};
#pragma unroll
            for (int c = 0; c < 64; ++c) {
                v2f bfrag = *(const v2f*)(w2t + (size_t)(n0 + mn)*DFF + 4*c + 2*half);
                acc = __builtin_amdgcn_wmma_f32_16x16x4_f32(false, areg[c], false, bfrag,
                                                            (short)0, acc, false, false);
            }
            const float bias = B2[n0 + mn];
#pragma unroll
            for (int r = 0; r < 8; ++r) {
                int m = mtile*16 + r + 8*half;
                yl[m*DM + n0 + mn] = acc[r] + bias + hl[m*DM + n0 + mn];
            }
        }
    }
    __syncthreads();

    // ---- LN2, 8 rows per wave.
    const int col = lane * 4;
    for (int rr = 0; rr < 8; ++rr) {
        const int m = wave*8 + rr;
        const float* yp = yl + m*DM + col;
        float y0 = yp[0], y1 = yp[1], y2 = yp[2], y3 = yp[3];
        float mu = wave_sum(y0 + y1 + y2 + y3) * (1.0f/128.0f);
        float d0 = y0 - mu, d1 = y1 - mu, d2 = y2 - mu, d3 = y3 - mu;
        float var = wave_sum(d0*d0 + d1*d1 + d2*d2 + d3*d3) * (1.0f/128.0f);
        float rs = __frsqrt_rn(var + 1e-5f);
        float* op = out + (row0 + m)*DM + col;
        op[0] = d0*rs*g2[col+0] + bln2[col+0];
        op[1] = d1*rs*g2[col+1] + bln2[col+1];
        op[2] = d2*rs*g2[col+2] + bln2[col+2];
        op[3] = d3*rs*g2[col+3] + bln2[col+3];
    }
}

extern "C" void kernel_launch(void* const* d_in, const int* in_sizes, int n_in,
                              void* d_out, int out_size, void* d_ws, size_t ws_size,
                              hipStream_t stream) {
    const float* x   = (const float*)d_in[0];
    const float* g1  = (const float*)d_in[1];
    const float* b1l = (const float*)d_in[2];
    const float* W1  = (const float*)d_in[3];
    const float* b1f = (const float*)d_in[4];
    const float* W2  = (const float*)d_in[5];
    const float* b2f = (const float*)d_in[6];
    const float* g2  = (const float*)d_in[7];
    const float* b2l = (const float*)d_in[8];
    float* out = (float*)d_out;

    // Workspace layout (bytes):
    //   h     : [0, 8 MB)                      f32  [B*S, D]
    //   xhi   : [8 MB, 12 MB)                  bf16 [B*S, D]
    //   xlo   : [12 MB, 16 MB)                 bf16 [B*S, D]
    //   w1t   : [16 MB, 16 MB + 128 KB)        f32  [DFF, D]   (W1 transposed)
    //   w2t   : [+128 KB, +256 KB)             f32  [D, DFF]   (W2 transposed)
    char* ws = (char*)d_ws;
    float*  hbuf = (float*)ws;
    __bf16* xhi  = (__bf16*)(ws + (size_t)8  * 1024 * 1024);
    __bf16* xlo  = (__bf16*)(ws + (size_t)12 * 1024 * 1024);
    float*  w1t  = (float*)(ws + (size_t)16 * 1024 * 1024);
    float*  w2t  = (float*)(ws + (size_t)16 * 1024 * 1024 + 128 * 1024);

    const int total = BATCH * SEQ * DM;
    prep_kernel<<<dim3((total + 255) / 256), 256, 0, stream>>>(x, W1, W2, xhi, xlo, w1t, w2t);

    const size_t lds1 = (size_t)16*SEQ*4 + 16*256*4 + 16*MAXK*4*2 + 16*4*2;
    attn_ln1_kernel<<<dim3(SEQ/16, BATCH), BLOCK1, lds1, stream>>>(x, xhi, xlo, g1, b1l, hbuf);

    const size_t lds2 = (size_t)(128*DM + 128*DFF + 128*DM) * 4;
    ffn_ln2_kernel<<<dim3((BATCH*SEQ)/128), BLOCK2, lds2, stream>>>(hbuf, w1t, b1f, w2t, b2f,
                                                                    g2, b2l, out);
}
